// TimesBlock_81252191306617
// MI455X (gfx1250) — matrix operand order repaired and checked by reading, not compiled, not run on hardware
//
#include <hip/hip_runtime.h>
#include <hip/hip_bf16.h>
#include <math.h>

typedef __attribute__((ext_vector_type(16))) _Float16 v16h;
typedef __attribute__((ext_vector_type(8)))  float    v8f;

#define B_    16
#define L_    1024
#define D_    128
#define D2_   256
#define NFREQ 513
#define NPAIR 48          // 16 batches x top-3 periods
#define NTILES 97         // ceil(max Tp (=1533) / 16) = 96, plus one spare tile
#define PCOLS (NTILES*16) // 1552 columns reserved per pair in h scratch
#define KT1   36          // 1152 / 32
#define KT2   72          // 2304 / 32

// dynamic LDS, 64B aligned so v16h (32B) loads are aligned
extern __shared__ __align__(64) _Float16 smemh[];

// ---------------------------------------------------------------------------
// 1) mean over channels: xm[b][l] = mean_d x[b][l][d]
__global__ void mean_kernel(const float* __restrict__ x, float* __restrict__ xm) {
    int i = blockIdx.x * 256 + threadIdx.x;
    if (i >= B_ * L_) return;
    const float* row = x + (size_t)i * D_;
    float s = 0.f;
    for (int d = 0; d < D_; ++d) s += row[d];
    xm[i] = s * (1.0f / D_);
}

// 2) |rfft| via brute-force DFT with exact integer phase (mod 1024)
__global__ void dft_kernel(const float* __restrict__ xm, float* __restrict__ mag) {
    int i = blockIdx.x * 256 + threadIdx.x;
    if (i >= B_ * NFREQ) return;
    int b = i / NFREQ, k = i % NFREQ;
    if (k == 0) { mag[i] = 0.f; return; }
    const float* row = xm + b * L_;
    float sr = 0.f, si = 0.f;
    int m = 0;
    const float step = -6.283185307179586f / (float)L_;
    for (int l = 0; l < L_; ++l) {
        float s, c;
        __sincosf(step * (float)m, &s, &c);
        float v = row[l];
        sr += v * c;
        si += v * s;
        m = (m + k) & (L_ - 1);
    }
    mag[i] = sqrtf(sr * sr + si * si);
}

// 3) per-batch top-3 bins -> (b, period) work items; period<=1 marked invalid
__global__ void top3_kernel(const float* __restrict__ mag, int2* __restrict__ pairs) {
    if (threadIdx.x != 0) return;
    int b = blockIdx.x;
    const float* m = mag + b * NFREQ;
    int sel0 = -1, sel1 = -1, sel2 = -1;
    for (int j = 0; j < 3; ++j) {
        float best = -1e30f; int bi = 0;
        for (int k = 0; k < NFREQ; ++k) {
            if (k == sel0 || k == sel1 || k == sel2) continue;
            float v = m[k];
            if (v > best) { best = v; bi = k; }   // ties -> lowest index (matches top_k)
        }
        if (j == 0) sel0 = bi; else if (j == 1) sel1 = bi; else sel2 = bi;
        int p = bi + 1;
        pairs[b * 3 + j] = make_int2(b, (p > 1) ? p : 0);
    }
}

// 4) zero the f32 accumulation buffer
__global__ void zero_kernel(float* __restrict__ buf, int n) {
    int i = blockIdx.x * 256 + threadIdx.x;
    if (i < n) buf[i] = 0.f;
}

// ---------------------------------------------------------------------------
// Weight packing into WMMA A-fragment order.
// A f16 16x32 per-lane order used here: M = lane&15,
// half j -> K = j + ((lane>=16) ? 16 : 0) (contiguous k; B LDS reads use the same).
// GEMM K ordering: k = (dt*3+dq)*C + c  (so im2col chunks are channel-contiguous)
__global__ void pack_w1(const float* __restrict__ w, _Float16* __restrict__ pa) {
    int idx = blockIdx.x * 256 + threadIdx.x;          // [mt][kt][lane][j]
    if (idx >= 16 * KT1 * 32 * 16) return;
    int j = idx & 15, r = idx >> 4;
    int lane = r & 31; int r2 = r >> 5;
    int kt = r2 % KT1, mt = r2 / KT1;
    int m = mt * 16 + (lane & 15);
    int k = kt * 32 + j + ((lane >> 4) << 4);
    int g = k >> 7, c = k & 127;                       // k = g*128 + c
    int dt = g / 3, dq = g % 3;
    pa[idx] = (_Float16)w[((m * D_ + c) * 3 + dt) * 3 + dq];
}
__global__ void pack_w2(const float* __restrict__ w, _Float16* __restrict__ pa) {
    int idx = blockIdx.x * 256 + threadIdx.x;
    if (idx >= 8 * KT2 * 32 * 16) return;
    int j = idx & 15, r = idx >> 4;
    int lane = r & 31; int r2 = r >> 5;
    int kt = r2 % KT2, mt = r2 / KT2;
    int m = mt * 16 + (lane & 15);
    int k = kt * 32 + j + ((lane >> 4) << 4);
    int g = k >> 8, c = k & 255;                       // k = g*256 + c
    int dt = g / 3, dq = g % 3;
    pa[idx] = (_Float16)w[((m * D2_ + c) * 3 + dt) * 3 + dq];
}

// ---------------------------------------------------------------------------
// conv1 (implicit GEMM, M=256, K=1152, N=Tp) + GELU, h -> f16 scratch [pair][n][256]
__global__ void conv1_kernel(const float* __restrict__ x,
                             const _Float16* __restrict__ pa1,
                             const float* __restrict__ b1,
                             const int2* __restrict__ pairs,
                             _Float16* __restrict__ hbuf) {
    const int CK = 3 * D_;                 // LDS K-chunk: 3 taps x 128 ch = 384 halves
    int pairIdx = blockIdx.x;
    int2 pr = pairs[pairIdx];
    int b = pr.x, p = pr.y;
    if (p <= 1) return;
    int T = (L_ + p - 1) / p;
    int N = T * p;
    int n0 = (int)blockIdx.y * 16;
    if (n0 >= N) return;

    int tid = threadIdx.x, wave = tid >> 5, lane = tid & 31;
    v8f acc0 = {}, acc1 = {};
    const _Float16* bptr = smemh + (size_t)(lane & 15) * CK + ((lane >> 4) << 4);

    for (int cg = 0; cg < 3; ++cg) {                 // chunk over the 9 taps, 3 at a time
        for (int e = tid; e < 16 * CK; e += 256) {   // stage im2col B tile [col][k]
            int col = e / CK, kk = e % CK;
            int g = cg * 3 + (kk >> 7);
            int c = kk & 127;
            int n = n0 + col;
            float v = 0.f;
            if (n < N) {
                int t = n / p, q = n - t * p;
                int tt = t + g / 3 - 1, qq = q + g % 3 - 1;
                if (tt >= 0 && tt < T && qq >= 0 && qq < p) {
                    int l = tt * p + qq;
                    if (l < L_) v = x[((size_t)b * L_ + l) * D_ + c];
                }
            }
            smemh[col * CK + kk] = (_Float16)v;
        }
        __syncthreads();
        for (int kt = 0; kt < CK / 32; ++kt) {       // 12 K-steps of 32
            int ktg = cg * (CK / 32) + kt;
            v16h bf = *(const v16h*)(bptr + kt * 32);
            v16h a0 = *(const v16h*)(pa1 + ((size_t)((wave)     * KT1 + ktg) * 32 + lane) * 16);
            v16h a1 = *(const v16h*)(pa1 + ((size_t)((wave + 8) * KT1 + ktg) * 32 + lane) * 16);
            acc0 = __builtin_amdgcn_wmma_f32_16x16x32_f16(false, a0, false, bf,
                                                          (short)0, acc0, false, false);
            acc1 = __builtin_amdgcn_wmma_f32_16x16x32_f16(false, a1, false, bf,
                                                          (short)0, acc1, false, false);
        }
        __syncthreads();
    }

    int n = n0 + (lane & 15);
    if (n < N) {
        int m0 = wave * 16 + ((lane >> 4) << 3);
        int m1 = (wave + 8) * 16 + ((lane >> 4) << 3);
        _Float16* dst = hbuf + ((size_t)pairIdx * PCOLS + n) * D2_;
        for (int v = 0; v < 8; ++v) {
            float h0 = acc0[v] + b1[m0 + v];
            h0 = 0.5f * h0 * (1.0f + erff(h0 * 0.70710678118654752f));   // exact GELU
            dst[m0 + v] = (_Float16)h0;
            float h1 = acc1[v] + b1[m1 + v];
            h1 = 0.5f * h1 * (1.0f + erff(h1 * 0.70710678118654752f));
            dst[m1 + v] = (_Float16)h1;
        }
    }
}

// conv2 (implicit GEMM, M=128, K=2304, N=Tp), atomically accumulate into obuf[b][l][c]
__global__ void conv2_kernel(const _Float16* __restrict__ hbuf,
                             const _Float16* __restrict__ pa2,
                             const float* __restrict__ b2,
                             const int2* __restrict__ pairs,
                             float* __restrict__ obuf) {
    const int CK = 3 * D2_;                // 3 taps x 256 ch = 768 halves per chunk
    int pairIdx = blockIdx.x;
    int2 pr = pairs[pairIdx];
    int b = pr.x, p = pr.y;
    if (p <= 1) return;
    int T = (L_ + p - 1) / p;
    int N = T * p;
    int n0 = (int)blockIdx.y * 16;
    if (n0 >= N) return;

    int tid = threadIdx.x, wave = tid >> 5, lane = tid & 31;
    v8f acc = {};
    const _Float16* bptr = smemh + (size_t)(lane & 15) * CK + ((lane >> 4) << 4);
    const _Float16* hsrc = hbuf + (size_t)pairIdx * PCOLS * D2_;

    for (int cg = 0; cg < 3; ++cg) {
        for (int e = tid; e < 16 * CK; e += 256) {
            int col = e / CK, kk = e % CK;
            int g = cg * 3 + (kk >> 8);
            int c = kk & 255;
            int n = n0 + col;
            float v = 0.f;
            if (n < N) {
                int t = n / p, q = n - t * p;
                int tt = t + g / 3 - 1, qq = q + g % 3 - 1;
                if (tt >= 0 && tt < T && qq >= 0 && qq < p)  // conv runs over padded grid
                    v = (float)hsrc[(size_t)(tt * p + qq) * D2_ + c];
            }
            smemh[col * CK + kk] = (_Float16)v;
        }
        __syncthreads();
        for (int kt = 0; kt < CK / 32; ++kt) {       // 24 K-steps of 32
            int ktg = cg * (CK / 32) + kt;
            v16h bf = *(const v16h*)(bptr + kt * 32);
            v16h a0 = *(const v16h*)(pa2 + ((size_t)(wave * KT2 + ktg) * 32 + lane) * 16);
            acc = __builtin_amdgcn_wmma_f32_16x16x32_f16(false, a0, false, bf,
                                                         (short)0, acc, false, false);
        }
        __syncthreads();
    }

    int n = n0 + (lane & 15);                        // n == l (row-major fold)
    if (n < L_) {                                    // drop padded tail
        int c0 = wave * 16 + ((lane >> 4) << 3);
        float* dst = obuf + ((size_t)b * L_ + n) * D_;
        for (int v = 0; v < 8; ++v)
            atomicAdd(&dst[c0 + v], acc[v] + b2[c0 + v]);
    }
}

// ---------------------------------------------------------------------------
// residual + /3 + LayerNorm(D=128)
__global__ void ln_kernel(const float* __restrict__ x, const float* __restrict__ obuf,
                          const float* __restrict__ g, const float* __restrict__ beta,
                          float* __restrict__ out) {
    __shared__ float red[D_];
    int row = blockIdx.x, d = threadIdx.x;
    size_t base = (size_t)row * D_ + d;
    float y = x[base] + obuf[base] * (1.0f / 3.0f);
    red[d] = y; __syncthreads();
    for (int s = 64; s > 0; s >>= 1) { if (d < s) red[d] += red[d + s]; __syncthreads(); }
    float mu = red[0] * (1.0f / D_); __syncthreads();
    float dy = y - mu;
    red[d] = dy * dy; __syncthreads();
    for (int s = 64; s > 0; s >>= 1) { if (d < s) red[d] += red[d + s]; __syncthreads(); }
    float var = red[0] * (1.0f / D_);
    out[base] = dy * rsqrtf(var + 1e-5f) * g[d] + beta[d];
}

// ---------------------------------------------------------------------------
extern "C" void kernel_launch(void* const* d_in, const int* in_sizes, int n_in,
                              void* d_out, int out_size, void* d_ws, size_t ws_size,
                              hipStream_t stream) {
    (void)in_sizes; (void)n_in; (void)out_size; (void)ws_size;
    const float* x   = (const float*)d_in[0];
    const float* w1  = (const float*)d_in[1];
    const float* b1  = (const float*)d_in[2];
    const float* w2  = (const float*)d_in[3];
    const float* b2  = (const float*)d_in[4];
    const float* lg  = (const float*)d_in[5];
    const float* lb  = (const float*)d_in[6];
    float* out = (float*)d_out;

    char* wsp = (char*)d_ws;
    auto alloc = [&](size_t bytes) {
        char* r = wsp; wsp += (bytes + 255) & ~(size_t)255; return r;
    };
    float*    xm    = (float*)alloc((size_t)B_ * L_ * 4);
    float*    mag   = (float*)alloc((size_t)B_ * NFREQ * 4);
    int2*     pairs = (int2*)alloc((size_t)NPAIR * sizeof(int2));
    _Float16* pa1   = (_Float16*)alloc((size_t)16 * KT1 * 512 * 2);
    _Float16* pa2   = (_Float16*)alloc((size_t)8 * KT2 * 512 * 2);
    float*    obuf  = (float*)alloc((size_t)B_ * L_ * D_ * 4);
    _Float16* hbuf  = (_Float16*)alloc((size_t)NPAIR * PCOLS * D2_ * 2);

    int nacc = B_ * L_ * D_;
    zero_kernel<<<(nacc + 255) / 256, 256, 0, stream>>>(obuf, nacc);
    mean_kernel<<<(B_ * L_ + 255) / 256, 256, 0, stream>>>(x, xm);
    dft_kernel<<<(B_ * NFREQ + 255) / 256, 256, 0, stream>>>(xm, mag);
    top3_kernel<<<B_, 32, 0, stream>>>(mag, pairs);
    pack_w1<<<(16 * KT1 * 512 + 255) / 256, 256, 0, stream>>>(w1, pa1);
    pack_w2<<<(8 * KT2 * 512 + 255) / 256, 256, 0, stream>>>(w2, pa2);

    dim3 cgrid(NPAIR, NTILES);
    conv1_kernel<<<cgrid, 256, 16 * 3 * D_  * sizeof(_Float16), stream>>>(x,    pa1, b1, pairs, hbuf);
    conv2_kernel<<<cgrid, 256, 16 * 3 * D2_ * sizeof(_Float16), stream>>>(hbuf, pa2, b2, pairs, obuf);

    ln_kernel<<<B_ * L_, D_, 0, stream>>>(x, obuf, lg, lb, out);
}
